// BuzCusStructureSim_52432960749711
// MI455X (gfx1250) — compile-verified, matrix operand order checked by reading, not run on hardware
//
#include <hip/hip_runtime.h>
#include <hip/hip_bf16.h>

// Problem constants (from reference)
constexpr int kB  = 16;
constexpr int kS  = 512;
constexpr int kD  = 256;
constexpr int kH  = 8;
constexpr int kFB = 128;
constexpr int kK  = 64;
constexpr int kFC = 32;

typedef __attribute__((ext_vector_type(16))) __bf16 bf16x16;
typedef __attribute__((ext_vector_type(8)))  float  v8f;
typedef unsigned int u32x4 __attribute__((ext_vector_type(4)));
typedef int          i32x4 __attribute__((ext_vector_type(4)));
typedef int          i32x8 __attribute__((ext_vector_type(8)));

__device__ __forceinline__ unsigned short f2bf(float x) {
    unsigned int u = __float_as_uint(x);
    unsigned int r = u + 0x7FFFu + ((u >> 16) & 1u);   // round-to-nearest-even
    return (unsigned short)(r >> 16);
}

__device__ __forceinline__ v8f wmma_bf16(bf16x16 a, bf16x16 b, v8f c) {
    return __builtin_amdgcn_wmma_f32_16x16x32_bf16(false, a, false, b, (short)0, c,
                                                   false, false);
}

// ---------------- f32 -> bf16 conversion ----------------
__global__ void k_cvt_bf16(const float* __restrict__ src,
                           unsigned short* __restrict__ dst, int n) {
    int i = blockIdx.x * blockDim.x + threadIdx.x;
    if (i < n) dst[i] = f2bf(src[i]);
}

// ---------------- MT[h][j][i] = (1/sqrt(D)) * dot(W1[h,i,:], W2[h,j,:]) ------
// stored j-major (transposed M) so WMMA B-operand reads are contiguous per lane
__global__ void k_build_MT(const float* __restrict__ W1,
                           const float* __restrict__ W2,
                           unsigned short* __restrict__ MT) {
    int t = blockIdx.x * blockDim.x + threadIdx.x;     // h*D*D + j*D + i
    int i = t & (kD - 1);
    int j = (t >> 8) & (kD - 1);
    int h = t >> 16;
    const float* r1 = W1 + ((size_t)h * kD + i) * kD;
    const float* r2 = W2 + ((size_t)h * kD + j) * kD;
    float s = 0.f;
#pragma unroll 4
    for (int e = 0; e < kD; e += 4) {
        float4 a = *(const float4*)(r1 + e);
        float4 b = *(const float4*)(r2 + e);
        s += a.x * b.x + a.y * b.y + a.z * b.z + a.w * b.w;
    }
    MT[t] = f2bf(s * 0.0625f);   // 1/sqrt(256)
}

// ---------------- Bvec[b,s] = sum_f B_x[b,s,f]  (softmax of [FB,1] == ones) --
__global__ void k_sumB(const float* __restrict__ Bt_in,
                       const float* __restrict__ Bi_in,
                       float* __restrict__ Bt, float* __restrict__ Bi) {
    int wave = blockIdx.x * (blockDim.x >> 5) + (threadIdx.x >> 5);
    int lane = threadIdx.x & 31;
    const float* src = (blockIdx.y == 0) ? Bt_in : Bi_in;
    float* dst       = (blockIdx.y == 0) ? Bt    : Bi;
    const float* row = src + (size_t)wave * kFB;
    float s = 0.f;
    for (int f = lane; f < kFB; f += 32) s += row[f];
#pragma unroll
    for (int off = 16; off; off >>= 1) s += __shfl_xor(s, off, 32);
    if (lane == 0) dst[wave] = s;
}

// ---------------- Cv[b,s] = 0.5*KLD(Ci,Cm)+0.5*KLD(Ct,Cm) over K ------------
__global__ void k_cv(const float* __restrict__ Ct_in,
                     const float* __restrict__ Ci_in,
                     float* __restrict__ Cv) {
    int bs = blockIdx.x;
    int k  = threadIdx.x;                      // 0..63
    const float* ct = Ct_in + ((size_t)bs * kK + k) * kFC;
    const float* ci = Ci_in + ((size_t)bs * kK + k) * kFC;
    float st = 0.f, si = 0.f;
#pragma unroll
    for (int f = 0; f < kFC; f += 4) {
        float4 a = *(const float4*)(ct + f);
        float4 b = *(const float4*)(ci + f);
        st += a.x + a.y + a.z + a.w;
        si += b.x + b.y + b.z + b.w;
    }
    float cm = 0.5f * (st + si);
    const float eps = 1e-7f;
    float yp = fminf(fmaxf(cm, eps), 1.0f);
    float yi = fminf(fmaxf(si, eps), 1.0f);
    float yt = fminf(fmaxf(st, eps), 1.0f);
    float v = 0.5f * (yi * __logf(yi / yp) + yt * __logf(yt / yp));
#pragma unroll
    for (int off = 16; off; off >>= 1) v += __shfl_xor(v, off, 32);
    __shared__ float sm[2];
    if ((threadIdx.x & 31) == 0) sm[threadIdx.x >> 5] = v;
    __syncthreads();
    if (threadIdx.x == 0) Cv[bs] = sm[0] + sm[1];
}

// ---------------- P[b,h,s,:] = Ebf[b,s,:] @ M_h  (bf16 WMMA GEMM) -----------
// block: 128 rows x 64 cols. B-operand tile (64x256 bf16 = 32 KB, contiguous
// slab of MT) is DMA'd into LDS once per block by the Tensor Data Mover, then
// all 8 waves feed WMMA B-operands from LDS. A-operand streams from global.
__global__ void k_gemm_P(const unsigned short* __restrict__ Ebf,  // [B,S,D]
                         const unsigned short* __restrict__ MT,   // [H,Dj,Di]
                         unsigned short* __restrict__ P) {        // [B,H,S,D]
    __shared__ unsigned short Btile[64 * kD];   // 32 KB
    int warp = threadIdx.x >> 5;
    int lane = threadIdx.x & 31;
    int jt = blockIdx.x & 3;           // D/64 column tiles
    int sb = (blockIdx.x >> 2) & 3;    // S/128 row blocks
    int h  = (blockIdx.x >> 4) & 7;
    int b  = blockIdx.x >> 7;

    // ---- TDM: 1-D tile of 16384 bf16 elements (32 KB contiguous) -> LDS ----
    if (warp == 0) {
        unsigned long long ga =
            (unsigned long long)(const void*)(MT + ((size_t)h * kD + jt * 64) * kD);
        unsigned int lds_base = (unsigned int)(size_t)(void*)&Btile[0];
        u32x4 g0 = (u32x4)0u;
        g0[0] = 1u;                                    // count=1 (valid), load
        g0[1] = lds_base;                              // lds_addr
        g0[2] = (unsigned int)(ga & 0xFFFFFFFFu);      // global_addr[31:0]
        g0[3] = (unsigned int)((ga >> 32) & 0x01FFFFFFu) | (2u << 30); // type=2
        i32x8 g1 = (i32x8)0;
        g1[0] = (1 << 16);          // data_size=1 (2 bytes); no mask/pad/iterate
        g1[1] = (int)(16384u << 16);        // tensor_dim0[15:0]=16384
        g1[2] = (int)(1u << 16);            // tensor_dim0 hi=0, tensor_dim1=1
        g1[3] = (int)(16384u << 16);        // tensor_dim1 hi=0, tile_dim0=16384
        g1[4] = 1;                          // tile_dim1=1, tile_dim2=0
        g1[5] = 16384;                      // tensor_dim0_stride[31:0]
        i32x4 g2 = (i32x4)0;
        i32x4 g3 = (i32x4)0;
        i32x8 g4 = (i32x8)0;                // extra group (clang-23 6-arg form)
        __builtin_amdgcn_tensor_load_to_lds(g0, g1, g2, g3, g4, 0);
        __builtin_amdgcn_s_wait_tensorcnt(0);
    }
    __syncthreads();

    int row   = lane & 15;
    int khalf = lane >> 4;
    const unsigned short* Arow =
        Ebf + ((size_t)b * kS + sb * 128 + warp * 16 + row) * kD + khalf * 16;
    const unsigned short* Bc = &Btile[(size_t)(lane & 15) * kD + khalf * 16];

    v8f a0 = {}, a1 = {}, a2 = {}, a3 = {};
#pragma unroll
    for (int k = 0; k < kD; k += 32) {
        bf16x16 a  = *(const bf16x16*)(Arow + k);
        bf16x16 b0 = *(const bf16x16*)(Bc + k);
        bf16x16 b1 = *(const bf16x16*)(Bc + 16 * kD + k);
        bf16x16 b2 = *(const bf16x16*)(Bc + 32 * kD + k);
        bf16x16 b3 = *(const bf16x16*)(Bc + 48 * kD + k);
        a0 = wmma_bf16(a, b0, a0);
        a1 = wmma_bf16(a, b1, a1);
        a2 = wmma_bf16(a, b2, a2);
        a3 = wmma_bf16(a, b3, a3);
    }
    // C/D layout: lane<16 -> col n=lane, rows 0..7; lane>=16 -> rows 8..15
#pragma unroll
    for (int r = 0; r < 8; r++) {
        int m = khalf * 8 + r;
        size_t base = (((size_t)b * kH + h) * kS + sb * 128 + warp * 16 + m) * kD
                      + jt * 64;
        int n = lane & 15;
        P[base + n]      = f2bf(a0[r]);
        P[base + 16 + n] = f2bf(a1[r]);
        P[base + 32 + n] = f2bf(a2[r]);
        P[base + 48 + n] = f2bf(a3[r]);
    }
}

// ---------------- streaming column-softmax attention ------------------------
// scores[s,t] = P[b,h,s,:] . E[b,t,:]   (scale folded into M)
// out[b,t,h]  = sum_s Bvec[b,s] * softmax_s(scores[s,t])
// A-operand (P tile) is double-buffered so global latency overlaps the
// 8-WMMA chain + softmax VALU work of the previous s-tile.
__global__ void k_attn(const unsigned short* __restrict__ P,    // [B,H,S,D]
                       const unsigned short* __restrict__ Ebf,  // [B,S,D]
                       const float* __restrict__ Bvec,          // [B,S]
                       float* __restrict__ BR) {                // [B,S,H]
    int wave = blockIdx.x * (blockDim.x >> 5) + (threadIdx.x >> 5);
    int lane = threadIdx.x & 31;
    int tt = wave & 31;          // S/16 t-tiles
    int h  = (wave >> 5) & 7;
    int b  = wave >> 8;
    int col   = lane & 15;
    int khalf = lane >> 4;

    // hoist B operand: E rows for this 16-column t-tile, full D=256
    const unsigned short* Erow =
        Ebf + ((size_t)b * kS + tt * 16 + col) * kD + khalf * 16;
    bf16x16 Bop[8];
#pragma unroll
    for (int kk = 0; kk < 8; kk++) Bop[kk] = *(const bf16x16*)(Erow + kk * 32);

    const unsigned short* Pbase = P + (((size_t)b * kH + h) * kS) * kD;
    const float* Bv = Bvec + (size_t)b * kS;

    // prefetch first A tile
    bf16x16 Acur[8], Anext[8];
    {
        const unsigned short* A0 = Pbase + (size_t)(lane & 15) * kD + khalf * 16;
#pragma unroll
        for (int kk = 0; kk < 8; kk++) Acur[kk] = *(const bf16x16*)(A0 + kk * 32);
    }

    float m = -3.0e38f, den = 0.f, num = 0.f;
    for (int s0 = 0; s0 < kS; s0 += 16) {
        // prefetch next s-tile while current one computes
        if (s0 + 16 < kS) {
            const unsigned short* An =
                Pbase + (size_t)(s0 + 16 + (lane & 15)) * kD + khalf * 16;
#pragma unroll
            for (int kk = 0; kk < 8; kk++)
                Anext[kk] = *(const bf16x16*)(An + kk * 32);
        }
        v8f acc = {};
#pragma unroll
        for (int kk = 0; kk < 8; kk++) acc = wmma_bf16(Acur[kk], Bop[kk], acc);

        // column max across the 16 rows (two half-wave row groups)
        float lm = acc[0];
#pragma unroll
        for (int r = 1; r < 8; r++) lm = fmaxf(lm, acc[r]);
        lm = fmaxf(lm, __shfl_xor(lm, 16, 32));
        float newm = fmaxf(m, lm);
        float corr = __expf(m - newm);
        den *= corr;
        num *= corr;
        int rbase = s0 + khalf * 8;
        float4 w0 = *(const float4*)(Bv + rbase);
        float4 w1 = *(const float4*)(Bv + rbase + 4);
        float wv[8] = {w0.x, w0.y, w0.z, w0.w, w1.x, w1.y, w1.z, w1.w};
        float pd = 0.f, pn = 0.f;
#pragma unroll
        for (int r = 0; r < 8; r++) {
            float e = __expf(acc[r] - newm);
            pd += e;
            pn += e * wv[r];
        }
        pd += __shfl_xor(pd, 16, 32);
        pn += __shfl_xor(pn, 16, 32);
        den += pd;
        num += pn;
        m = newm;
        if (s0 + 16 < kS) {
#pragma unroll
            for (int kk = 0; kk < 8; kk++) Acur[kk] = Anext[kk];
        }
    }
    if (lane < 16)
        BR[((size_t)b * kS + tt * 16 + col) * kH + h] = num / den;
}

// ---------------- cos[b,s] = -dot(l2n(BR_i), l2n(BR_t)) over H --------------
__global__ void k_cos(const float* __restrict__ BRt,
                      const float* __restrict__ BRi,
                      float* __restrict__ cosv) {
    int bs = blockIdx.x * blockDim.x + threadIdx.x;
    if (bs >= kB * kS) return;
    const float* t = BRt + (size_t)bs * kH;
    const float* i_ = BRi + (size_t)bs * kH;
    float st = 0.f, si = 0.f, sd = 0.f;
    float tv[kH], iv[kH];
#pragma unroll
    for (int h = 0; h < kH; h++) {
        tv[h] = t[h];
        iv[h] = i_[h];
        st += tv[h] * tv[h];
        si += iv[h] * iv[h];
    }
    float rt = rsqrtf(fmaxf(st, 1e-12f));
    float ri = rsqrtf(fmaxf(si, 1e-12f));
#pragma unroll
    for (int h = 0; h < kH; h++) sd += (tv[h] * rt) * (iv[h] * ri);
    cosv[bs] = -sd;
}

// ---------------- layernorm over S (per batch) + final combine --------------
__device__ __forceinline__ float block_sum_512(float v, float* red) {
#pragma unroll
    for (int off = 16; off; off >>= 1) v += __shfl_xor(v, off, 32);
    int w = threadIdx.x >> 5, l = threadIdx.x & 31;
    __syncthreads();
    if (l == 0) red[w] = v;
    __syncthreads();
    float t = 0.f;
#pragma unroll
    for (int i = 0; i < 16; i++) t += red[i];
    return t;
}

__global__ void k_final(const float* __restrict__ cosv,
                        const float* __restrict__ Cv,
                        const float* __restrict__ gbs, const float* __restrict__ bbs,
                        const float* __restrict__ gcs, const float* __restrict__ bcs,
                        float* __restrict__ out) {
    __shared__ float red[16];
    int b = blockIdx.x;
    int s = threadIdx.x;   // 512
    float x = cosv[(size_t)b * kS + s];
    float y = Cv[(size_t)b * kS + s];
    const float invS = 1.0f / (float)kS;
    float mx = block_sum_512(x, red) * invS;
    float vx = block_sum_512((x - mx) * (x - mx), red) * invS;
    float my = block_sum_512(y, red) * invS;
    float vy = block_sum_512((y - my) * (y - my), red) * invS;
    float bs_out = (x - mx) * rsqrtf(vx + 1e-16f) * gbs[s] + bbs[s];
    float cs_out = (y - my) * rsqrtf(vy + 1e-16f) * gcs[s] + bcs[s];
    size_t idx = (size_t)b * kS + s;
    // softmax(W_f [2,1], axis=-1) == ones -> final = BS + CS
    out[idx]               = bs_out + cs_out;
    out[kB * kS + idx]     = bs_out;
    out[2 * kB * kS + idx] = cs_out;
}

extern "C" void kernel_launch(void* const* d_in, const int* in_sizes, int n_in,
                              void* d_out, int out_size, void* d_ws, size_t ws_size,
                              hipStream_t stream) {
    (void)in_sizes; (void)n_in; (void)out_size; (void)ws_size;
    const float* B_target   = (const float*)d_in[0];
    const float* B_infected = (const float*)d_in[1];
    const float* E_target   = (const float*)d_in[2];
    const float* E_infected = (const float*)d_in[3];
    const float* C_target   = (const float*)d_in[4];
    const float* C_infected = (const float*)d_in[5];
    const float* W1         = (const float*)d_in[8];
    const float* W2         = (const float*)d_in[9];
    const float* gamma_bs   = (const float*)d_in[10];
    const float* beta_bs    = (const float*)d_in[11];
    const float* gamma_cs   = (const float*)d_in[12];
    const float* beta_cs    = (const float*)d_in[13];
    float* out = (float*)d_out;

    // workspace layout (256-byte aligned slices)
    char*  w   = (char*)d_ws;
    size_t off = 0;
    auto alloc = [&](size_t bytes) -> void* {
        void* p = w + off;
        off = (off + bytes + 255) & ~(size_t)255;
        return p;
    };
    unsigned short* Ebf_t = (unsigned short*)alloc((size_t)kB * kS * kD * 2);
    unsigned short* Ebf_i = (unsigned short*)alloc((size_t)kB * kS * kD * 2);
    unsigned short* MT    = (unsigned short*)alloc((size_t)kH * kD * kD * 2);
    unsigned short* P_t   = (unsigned short*)alloc((size_t)kB * kH * kS * kD * 2);
    unsigned short* P_i   = (unsigned short*)alloc((size_t)kB * kH * kS * kD * 2);
    float* Bt   = (float*)alloc((size_t)kB * kS * 4);
    float* Bi   = (float*)alloc((size_t)kB * kS * 4);
    float* BRt  = (float*)alloc((size_t)kB * kS * kH * 4);
    float* BRi  = (float*)alloc((size_t)kB * kS * kH * 4);
    float* cosv = (float*)alloc((size_t)kB * kS * 4);
    float* Cv   = (float*)alloc((size_t)kB * kS * 4);

    const int nE = kB * kS * kD;   // 2,097,152
    k_cvt_bf16<<<nE / 256, 256, 0, stream>>>(E_target, Ebf_t, nE);
    k_cvt_bf16<<<nE / 256, 256, 0, stream>>>(E_infected, Ebf_i, nE);
    k_build_MT<<<(kH * kD * kD) / 256, 256, 0, stream>>>(W1, W2, MT);
    k_sumB<<<dim3(kB * kS / 8, 2), 256, 0, stream>>>(B_target, B_infected, Bt, Bi);
    k_cv<<<kB * kS, kK, 0, stream>>>(C_target, C_infected, Cv);

    // GEMM P: blocks = B*H*(S/128)*(D/64) = 2048 of 8 waves (TDM-staged B tile)
    k_gemm_P<<<2048, 256, 0, stream>>>(Ebf_t, MT, P_t);
    k_gemm_P<<<2048, 256, 0, stream>>>(Ebf_i, MT, P_i);

    // attention: waves = B*H*(S/16) = 4096 -> 512 blocks of 8 waves
    k_attn<<<512, 256, 0, stream>>>(P_t, Ebf_t, Bt, BRt);
    k_attn<<<512, 256, 0, stream>>>(P_i, Ebf_i, Bi, BRi);

    k_cos<<<(kB * kS) / 256, 256, 0, stream>>>(BRt, BRi, cosv);
    k_final<<<kB, kS, 0, stream>>>(cosv, Cv, gamma_bs, beta_bs, gamma_cs, beta_cs, out);
}
